// Attention_37263136260335
// MI455X (gfx1250) — matrix operand order repaired and checked by reading, not compiled, not run on hardware
//
#include <hip/hip_runtime.h>
#include <hip/hip_bf16.h>
#include <math.h>

#define BATCH 8
#define SEQ   4096
#define CDIM  512
#define HEADS 8
#define HDIM  64
#define COUT  1536   // q(512) | k(512) | v(512)
#define EPSV  1e-6f

typedef __attribute__((ext_vector_type(16))) __bf16 v16bf;
typedef __attribute__((ext_vector_type(2)))  __bf16 v2bf;
typedef __attribute__((ext_vector_type(8)))  float  v8f;

union FragB16 {
  v16bf v;
  uint4 q[2];
};

// Native bf16 converts (v_cvt_bf16_f32 / v_cvt_pk_bf16_f32) instead of manual
// integer RNE — big VALU reduction in the conversion-heavy epilogues.
static __device__ __forceinline__ unsigned short f2bf(float f) {
  return __builtin_bit_cast(unsigned short, (__bf16)f);
}
static __device__ __forceinline__ unsigned pk2bf(float a, float b) {
  v2bf p;
  p.x = (__bf16)a;
  p.y = (__bf16)b;
  return __builtin_bit_cast(unsigned, p);
}
static __device__ __forceinline__ float bf2f(unsigned h) {
  return __uint_as_float(h << 16);
}
static __device__ __forceinline__ v8f v8f_zero() {
  v8f z = {0.f, 0.f, 0.f, 0.f, 0.f, 0.f, 0.f, 0.f};
  return z;
}

// Async global->LDS (ASYNCcnt) path, guarded so a missing builtin degrades to
// a plain copy instead of breaking the build. The builtin signature (per this
// toolchain's diagnostic) is:
//   void __builtin_amdgcn_global_load_async_to_lds_b128(
//       int4vec addrspace(1)* src, int4vec addrspace(3)* dst,
//       imm int offset, imm int cpol)
#if defined(__has_builtin)
#  if __has_builtin(__builtin_amdgcn_global_load_async_to_lds_b128) && \
      __has_builtin(__builtin_amdgcn_s_wait_asynccnt)
#    define USE_ASYNC_LDS 1
#  endif
#endif
#ifndef USE_ASYNC_LDS
#  define USE_ASYNC_LDS 0
#endif

typedef int async_int4 __attribute__((vector_size(16)));
#define AS1_V4I(p) ((__attribute__((address_space(1))) async_int4*)(p))
#define AS3_V4I(p) ((__attribute__((address_space(3))) async_int4*)(p))

// ---------------------------------------------------------------------------
// Kernel 0: convert Wq (512x512) + Wkv (1024x512) fp32 -> combined bf16
// weight matrix Wb (1536x512, row-major). 1.5 MB: L2-resident afterwards.
// ---------------------------------------------------------------------------
__global__ void k_convert_w(const float* __restrict__ Wq,
                            const float* __restrict__ Wkv,
                            unsigned short* __restrict__ Wb) {
  int idx = blockIdx.x * blockDim.x + threadIdx.x;
  if (idx >= COUT * CDIM) return;
  int row = idx / CDIM, col = idx - row * CDIM;
  float v = (row < CDIM) ? Wq[(size_t)row * CDIM + col]
                         : Wkv[(size_t)(row - CDIM) * CDIM + col];
  Wb[idx] = f2bf(v);
}

// ---------------------------------------------------------------------------
// Kernel 1: fused QKV projection. (B*N x 1536) = x @ Wb^T via
// v_wmma_f32_16x16x32_bf16; applies elu+1 to q/k and writes:
//   Qb: [b][n][512]        bf16 (A-layout for kernel 3)
//   KT: [b][h][64][4096]   bf16 (transposed -> A-layout for kernel 2)
//   VT: [b][h][64][4096]   bf16 (transposed -> B-layout for kernel 2)
// Block: 256 thr (8 waves), tile 128x128; wave = 64x32.
// x tile fp32->bf16 staged into double-buffered LDS (stride 80B, conflict-free)
// so only ONE barrier per K-step is needed.
// ---------------------------------------------------------------------------
__global__ __launch_bounds__(256) void k_qkv(
    const float* __restrict__ x,
    const unsigned short* __restrict__ Wb,
    unsigned short* __restrict__ Qb,
    unsigned short* __restrict__ KT,
    unsigned short* __restrict__ VT) {
  __shared__ unsigned short sA[2][128 * 40];
  const int tid  = threadIdx.x;
  const int lane = tid & 31;
  const int w    = tid >> 5;
  const int wm   = w >> 2;   // 0..1 -> row half (64 rows)
  const int wn   = w & 3;    // 0..3 -> col quarter (32 cols)
  const int rowBase = blockIdx.x * 128;   // global row over B*N
  const int colBase = blockIdx.y * 128;

  const int Kb   = (lane < 16) ? 0 : 8;    // A-fragment K base
  const int kb16 = (lane < 16) ? 0 : 16;   // unused; B fragment also uses Kb

  v8f acc[4][2];
#pragma unroll
  for (int i = 0; i < 4; ++i)
#pragma unroll
    for (int j = 0; j < 2; ++j) acc[i][j] = v8f_zero();

  int buf = 0;
  for (int kk = 0; kk < CDIM; kk += 32) {
#pragma unroll
    for (int p = 0; p < 4; ++p) {
      int r  = p * 32 + (tid >> 3);
      int c4 = (tid & 7) * 4;
      const float* xp = &x[(size_t)(rowBase + r) * CDIM + kk + c4];
      const float4 xv = *(const float4*)xp;
      if (kk + 32 < CDIM) __builtin_prefetch(xp + 32, 0, 0);
      *(uint2*)&sA[buf][r * 40 + c4] =
          make_uint2(pk2bf(xv.x, xv.y), pk2bf(xv.z, xv.w));
    }
    __syncthreads();

    FragB16 bfr[2];
#pragma unroll
    for (int ni = 0; ni < 2; ++ni) {
      int j = colBase + wn * 32 + ni * 16 + (lane & 15);
      const unsigned short* bp = &Wb[(size_t)j * CDIM + kk + Kb];
      bfr[ni].q[0] = *(const uint4*)bp;
      bfr[ni].q[1] = *(const uint4*)(bp + 16);
    }
#pragma unroll
    for (int mi = 0; mi < 4; ++mi) {
      int r = wm * 64 + mi * 16 + (lane & 15);
      FragB16 afr;
      const unsigned short* ap = &sA[buf][r * 40 + Kb];
      afr.q[0] = *(const uint4*)ap;
      afr.q[1] = *(const uint4*)(ap + 16);   // K = 16+Kb .. 16+Kb+7
#pragma unroll
      for (int ni = 0; ni < 2; ++ni) {
        acc[mi][ni] = __builtin_amdgcn_wmma_f32_16x16x32_bf16(
            false, afr.v, false, bfr[ni].v, (short)0, acc[mi][ni], false,
            false);
      }
    }
    buf ^= 1;
  }

  // Epilogue: acc element (r,lane) -> (M = r + (lane>=16?8:0), Ncol = lane%16)
#pragma unroll
  for (int mi = 0; mi < 4; ++mi) {
#pragma unroll
    for (int ni = 0; ni < 2; ++ni) {
      int col = colBase + wn * 32 + ni * 16 + (lane & 15);
      if (col < CDIM) {                       // Q: elu+1, row-major store
#pragma unroll
        for (int r = 0; r < 8; ++r) {
          int lrow = wm * 64 + mi * 16 + r + ((lane >> 4) << 3);
          int g = rowBase + lrow;
          float val = acc[mi][ni][r];
          float a = (val > 0.f) ? (val + 1.f) : __expf(val);
          Qb[(size_t)g * CDIM + col] = f2bf(a);
        }
      } else if (col < 2 * CDIM) {            // K: elu+1, transposed store
        int jd = col - CDIM;
        int hh = jd >> 6, dd = jd & 63;
#pragma unroll
        for (int r = 0; r < 8; r += 2) {      // consecutive r == consecutive n
          int lrow = wm * 64 + mi * 16 + r + ((lane >> 4) << 3);
          int g = rowBase + lrow;
          int b = g >> 12, n = g & (SEQ - 1);
          float v0 = acc[mi][ni][r], v1 = acc[mi][ni][r + 1];
          float a0 = (v0 > 0.f) ? (v0 + 1.f) : __expf(v0);
          float a1 = (v1 > 0.f) ? (v1 + 1.f) : __expf(v1);
          *(unsigned*)&KT[(((size_t)b * HEADS + hh) * HDIM + dd) * SEQ + n] =
              pk2bf(a0, a1);
        }
      } else {                                // V: raw (1/N cancels with *N)
        int jd = col - 2 * CDIM;
        int hh = jd >> 6, dd = jd & 63;
#pragma unroll
        for (int r = 0; r < 8; r += 2) {
          int lrow = wm * 64 + mi * 16 + r + ((lane >> 4) << 3);
          int g = rowBase + lrow;
          int b = g >> 12, n = g & (SEQ - 1);
          *(unsigned*)&VT[(((size_t)b * HEADS + hh) * HDIM + dd) * SEQ + n] =
              pk2bf(acc[mi][ni][r], acc[mi][ni][r + 1]);
        }
      }
    }
  }
}

// ---------------------------------------------------------------------------
// Kernel 2: per (b,h): KV[dd][dv] = sum_s K[s][dd]*V[s][dv]  (64x64) via WMMA
// over N=4096; cross-wave reduce via ds_add_f32 LDS atomics. Writes KVT bf16
// transposed ([dv][dd] -> B-fragment layout for kernel 3) and Ksum[dd] f32.
// ---------------------------------------------------------------------------
__global__ __launch_bounds__(256) void k_kv(
    const unsigned short* __restrict__ KT,
    const unsigned short* __restrict__ VT,
    unsigned short* __restrict__ KVT,
    float* __restrict__ Ksum) {
  __shared__ float sKV[HDIM * HDIM];
  __shared__ float sKs[256];
  const int bh   = blockIdx.x;
  const int tid  = threadIdx.x;
  const int lane = tid & 31;
  const int w    = tid >> 5;

  for (int i = tid; i < HDIM * HDIM; i += 256) sKV[i] = 0.f;
  __syncthreads();

  const unsigned short* Kp = KT + (size_t)bh * HDIM * SEQ;
  const unsigned short* Vp = VT + (size_t)bh * HDIM * SEQ;
  const int Kb   = (lane < 16) ? 0 : 8;
  const int kb16 = (lane < 16) ? 0 : 16;

  v8f acc[4][4];
#pragma unroll
  for (int i = 0; i < 4; ++i)
#pragma unroll
    for (int j = 0; j < 4; ++j) acc[i][j] = v8f_zero();

  for (int s0 = w * 32; s0 < SEQ; s0 += 256) {
    FragB16 afr[4], bfr[4];
#pragma unroll
    for (int di = 0; di < 4; ++di) {           // A = K^T rows (dd, s)
      int dd = di * 16 + (lane & 15);
      const unsigned short* ap = &Kp[(size_t)dd * SEQ + s0 + Kb];
      afr[di].q[0] = *(const uint4*)ap;
      afr[di].q[1] = *(const uint4*)(ap + 16);
    }
#pragma unroll
    for (int vi = 0; vi < 4; ++vi) {           // B cols (s, dv) from VT rows
      int dv = vi * 16 + (lane & 15);
      const unsigned short* bp = &Vp[(size_t)dv * SEQ + s0 + Kb];
      bfr[vi].q[0] = *(const uint4*)bp;
      bfr[vi].q[1] = *(const uint4*)(bp + 16);
    }
#pragma unroll
    for (int di = 0; di < 4; ++di)
#pragma unroll
      for (int vi = 0; vi < 4; ++vi)
        acc[di][vi] = __builtin_amdgcn_wmma_f32_16x16x32_bf16(
            false, afr[di].v, false, bfr[vi].v, (short)0, acc[di][vi], false,
            false);
  }

  // cross-wave reduction into LDS (ds_add_f32)
#pragma unroll
  for (int di = 0; di < 4; ++di)
#pragma unroll
    for (int vi = 0; vi < 4; ++vi)
#pragma unroll
      for (int r = 0; r < 8; ++r) {
        int dd = di * 16 + r + ((lane >> 4) << 3);
        int dv = vi * 16 + (lane & 15);
        atomicAdd(&sKV[dd * HDIM + dv], acc[di][vi][r]);
      }
  __syncthreads();

  // KVT[dv][dd] bf16 (conflict-free LDS reads, scattered tiny stores)
  for (int e = tid; e < HDIM * HDIM; e += 256) {
    int dd = e >> 6, dv = e & 63;
    KVT[(size_t)bh * HDIM * HDIM + dv * HDIM + dd] = f2bf(sKV[e]);
  }

  // Ksum[dd] = sum_n K[n][dd]
  {
    int dd = tid & 63, seg = tid >> 6;
    const unsigned short* kp = &Kp[(size_t)dd * SEQ + seg * 1024];
    float s = 0.f;
    for (int i = 0; i < 1024; i += 8) {
      uint4 qv = *(const uint4*)&kp[i];
      s += bf2f(qv.x & 0xFFFFu) + bf2f(qv.x >> 16);
      s += bf2f(qv.y & 0xFFFFu) + bf2f(qv.y >> 16);
      s += bf2f(qv.z & 0xFFFFu) + bf2f(qv.z >> 16);
      s += bf2f(qv.w & 0xFFFFu) + bf2f(qv.w >> 16);
    }
    sKs[tid] = s;
  }
  __syncthreads();
  if (tid < HDIM)
    Ksum[(size_t)bh * HDIM + tid] =
        sKs[tid] + sKs[64 + tid] + sKs[128 + tid] + sKs[192 + tid];
}

// ---------------------------------------------------------------------------
// Kernel 3: out[n][dv] = (Q[n]·KV[:,dv]) * Z[n], Z[n] = 1/(Q[n]·Ksum + eps).
// KV tile copied global->LDS with GLOBAL_LOAD_ASYNC_TO_LDS_B128 when
// available (ASYNCcnt path); Ksum + per-row Z in LDS.
// Block: 256 thr, 256 rows of one (b,h); wave = 32 rows x 64 cols.
// ---------------------------------------------------------------------------
__global__ __launch_bounds__(256) void k_out(
    const unsigned short* __restrict__ Qb,
    const unsigned short* __restrict__ KVT,
    const float* __restrict__ Ksum,
    float* __restrict__ out) {
  __shared__ unsigned short sKV[HDIM * HDIM];  // [dv][dd]
  __shared__ float sKs[HDIM];
  __shared__ float sZ[256];
  const int rt = blockIdx.x, hh = blockIdx.y, b = blockIdx.z;
  const int bh = b * HEADS + hh;
  const int tid  = threadIdx.x;
  const int lane = tid & 31;
  const int w    = tid >> 5;
  const int rowBase = rt * 256;

  const unsigned short* gkv = KVT + (size_t)bh * HDIM * HDIM;
#if USE_ASYNC_LDS
#pragma unroll
  for (int it = 0; it < 2; ++it) {
    int c = (tid + it * 256) * 8;   // 8 bf16 = 16B per transfer
    __builtin_amdgcn_global_load_async_to_lds_b128(
        AS1_V4I(gkv + c), AS3_V4I(&sKV[c]), 0, 0);
  }
  __builtin_amdgcn_s_wait_asynccnt(0);
#else
  for (int i = tid; i < (HDIM * HDIM) / 2; i += 256)
    ((unsigned*)sKV)[i] = ((const unsigned*)gkv)[i];
#endif
  if (tid < HDIM) sKs[tid] = Ksum[(size_t)bh * HDIM + tid];
  __syncthreads();

  {  // Z: one row per lane (wave covers rows [w*32, w*32+32))
    int n = rowBase + w * 32 + lane;
    const unsigned short* qp =
        &Qb[((size_t)(b * SEQ + n)) * CDIM + hh * HDIM];
    float dot = 0.f;
#pragma unroll
    for (int d0 = 0; d0 < HDIM; d0 += 8) {
      uint4 qv = *(const uint4*)&qp[d0];
      dot += bf2f(qv.x & 0xFFFFu) * sKs[d0 + 0] + bf2f(qv.x >> 16) * sKs[d0 + 1];
      dot += bf2f(qv.y & 0xFFFFu) * sKs[d0 + 2] + bf2f(qv.y >> 16) * sKs[d0 + 3];
      dot += bf2f(qv.z & 0xFFFFu) * sKs[d0 + 4] + bf2f(qv.z >> 16) * sKs[d0 + 5];
      dot += bf2f(qv.w & 0xFFFFu) * sKs[d0 + 6] + bf2f(qv.w >> 16) * sKs[d0 + 7];
    }
    sZ[tid] = 1.f / (dot + EPSV);
  }
  __syncthreads();

  const int Kb   = (lane < 16) ? 0 : 8;
  const int kb16 = (lane < 16) ? 0 : 16;
  v8f acc[2][4];
#pragma unroll
  for (int i = 0; i < 2; ++i)
#pragma unroll
    for (int j = 0; j < 4; ++j) acc[i][j] = v8f_zero();

#pragma unroll
  for (int kk = 0; kk < HDIM; kk += 32) {
    FragB16 afr[2], bfr[4];
#pragma unroll
    for (int mi = 0; mi < 2; ++mi) {
      int n = rowBase + w * 32 + mi * 16 + (lane & 15);
      const unsigned short* ap =
          &Qb[((size_t)(b * SEQ + n)) * CDIM + hh * HDIM + kk + Kb];
      afr[mi].q[0] = *(const uint4*)ap;
      afr[mi].q[1] = *(const uint4*)(ap + 16);
    }
#pragma unroll
    for (int ni = 0; ni < 4; ++ni) {
      int dv = ni * 16 + (lane & 15);
      const unsigned short* bp = &sKV[dv * HDIM + kk + Kb];
      bfr[ni].q[0] = *(const uint4*)bp;
      bfr[ni].q[1] = *(const uint4*)(bp + 16);
    }
#pragma unroll
    for (int mi = 0; mi < 2; ++mi)
#pragma unroll
      for (int ni = 0; ni < 4; ++ni)
        acc[mi][ni] = __builtin_amdgcn_wmma_f32_16x16x32_bf16(
            false, afr[mi].v, false, bfr[ni].v, (short)0, acc[mi][ni], false,
            false);
  }

#pragma unroll
  for (int mi = 0; mi < 2; ++mi)
#pragma unroll
    for (int ni = 0; ni < 4; ++ni)
#pragma unroll
      for (int r = 0; r < 8; ++r) {
        int lr = w * 32 + mi * 16 + r + ((lane >> 4) << 3);
        int n = rowBase + lr;
        int dv = ni * 16 + (lane & 15);
        out[((size_t)(b * SEQ + n)) * CDIM + hh * HDIM + dv] =
            acc[mi][ni][r] * sZ[lr];
      }
}

// ---------------------------------------------------------------------------
// Workspace layout (bytes), ~103 MB total:
//   Wb  @ 0          : 1536*512*2       = 1,572,864
//   Qb  @ 1572864    : 8*4096*512*2     = 33,554,432
//   KT  @ +33554432  : 33,554,432
//   VT  @ +33554432  : 33,554,432
//   KVT @ +33554432  : 8*8*64*64*2      = 524,288
//   Ksum@ +524288    : 8*8*64*4         = 131,072
// ---------------------------------------------------------------------------
extern "C" void kernel_launch(void* const* d_in, const int* in_sizes, int n_in,
                              void* d_out, int out_size, void* d_ws,
                              size_t ws_size, hipStream_t stream) {
  (void)in_sizes; (void)n_in; (void)out_size; (void)ws_size;
  const float* x   = (const float*)d_in[0];
  const float* Wq  = (const float*)d_in[1];
  const float* Wkv = (const float*)d_in[2];
  float* out = (float*)d_out;

  char* ws = (char*)d_ws;
  unsigned short* Wb  = (unsigned short*)(ws);
  unsigned short* Qb  = (unsigned short*)(ws + 1572864);
  unsigned short* KT  = (unsigned short*)(ws + 1572864 + 1ull * 33554432);
  unsigned short* VT  = (unsigned short*)(ws + 1572864 + 2ull * 33554432);
  unsigned short* KVT = (unsigned short*)(ws + 1572864 + 3ull * 33554432);
  float* Ksum = (float*)(ws + 1572864 + 3ull * 33554432 + 524288);

  k_convert_w<<<(COUT * CDIM + 255) / 256, 256, 0, stream>>>(Wq, Wkv, Wb);
  k_qkv<<<dim3((BATCH * SEQ) / 128, COUT / 128), 256, 0, stream>>>(x, Wb, Qb,
                                                                   KT, VT);
  k_kv<<<BATCH * HEADS, 256, 0, stream>>>(KT, VT, KVT, Ksum);
  k_out<<<dim3(SEQ / 256, HEADS, BATCH), 256, 0, stream>>>(Qb, KVT, Ksum, out);
}
